// RoutingCapsules_7627861918151
// MI455X (gfx1250) — matrix operand order repaired and checked by reading, not compiled, not run on hardware
//
#include <hip/hip_runtime.h>
#include <math.h>

// ---------------- problem constants ----------------
#define CIN    2048
#define BTOT   128
#define NC     32
#define DC     32
#define DIN    32
#define ND     (NC*DC)          // 1024 combined (n,d) rows
#define BT     32               // batch tile per workgroup
#define CCHUNK 16               // channels per workgroup
#define CBLKS  (CIN/CCHUNK)     // 128

typedef __attribute__((ext_vector_type(16))) _Float16 v16h;
typedef __attribute__((ext_vector_type(8)))  float    v8f;

// ---------------- LDS layout (bytes), double-buffered staging ----------------
#define W0_OFF     0                      // [1024][32] f16 = 65536 B
#define W1_OFF     65536
#define X0_OFF     131072                 // [32][32] f16 = 2048 B
#define X1_OFF     133120
#define A_LDS_OFF  135168                 // [32 n][32 b] f32 logits = 4096 B
#define CF_LDS_OFF 139264                 // [32 n][32 b] f32 coef   = 4096 B
#define S_LDS_OFF  143360                 // [1024 m][33] f32 padded = 135168 B
#define S_STRIDE   33
#define SMEM_BYTES (S_LDS_OFF + ND*S_STRIDE*4)   // 278528 B < 320KB/WGP

// ============================================================
// W[1,n,c,d,i] f32  ->  Wh[c][m=n*32+d][i] f16  (channel-contiguous 64KB blocks)
// ============================================================
__global__ void caps_convert_W(const float* __restrict__ W, unsigned* __restrict__ Wh,
                               int total_dw) {
  for (int o = blockIdx.x * blockDim.x + threadIdx.x; o < total_dw;
       o += gridDim.x * blockDim.x) {
    int k2 = o & 15;              // f16 pair index (i = 2*k2)
    int m  = (o >> 4) & (ND - 1);
    int c  = o >> 14;
    int n = m >> 5, d = m & 31;
    size_t in = ((((size_t)n * CIN + c) * DC + d) * DIN) + 2 * k2;
    float2 f = *(const float2*)(W + in);
    union { _Float16 h[2]; unsigned u; } p;
    p.h[0] = (_Float16)f.x; p.h[1] = (_Float16)f.y;
    Wh[o] = p.u;
  }
}

// x[b,c,i] f32 -> xh[c][b][i] f16
__global__ void caps_convert_x(const float* __restrict__ x, unsigned* __restrict__ xh,
                               int total_dw) {
  for (int o = blockIdx.x * blockDim.x + threadIdx.x; o < total_dw;
       o += gridDim.x * blockDim.x) {
    int k2 = o & 15;
    int b  = (o >> 4) & (BTOT - 1);
    int c  = o >> 11;
    size_t in = (((size_t)b * CIN + c) * DIN) + 2 * k2;
    float2 f = *(const float2*)(x + in);
    union { _Float16 h[2]; unsigned u; } p;
    p.h[0] = (_Float16)f.x; p.h[1] = (_Float16)f.y;
    xh[o] = p.u;
  }
}

// ============================================================
// Async staging: every thread issues exactly 17 async loads (16 W b128 + 1 x b64)
// so the per-wave ASYNCcnt increment is uniform (17) across all 8 waves.
// ============================================================
__device__ __forceinline__ void stage_async(const _Float16* __restrict__ Wh,
                                            const _Float16* __restrict__ xh,
                                            void* wbuf, void* xbuf,
                                            int c, int b0, int tid) {
  // generic->LDS: low 32 bits of the generic address are the LDS byte offset
  unsigned wdst = (unsigned)(uintptr_t)wbuf + (unsigned)tid * 16u;
  unsigned wsrc = (unsigned)c * 65536u + (unsigned)tid * 16u;   // bytes into Wh
  #pragma unroll
  for (int i = 0; i < 16; ++i) {
    asm volatile("global_load_async_to_lds_b128 %0, %1, %2"
                 :: "v"(wdst + (unsigned)i * 4096u),
                    "v"(wsrc + (unsigned)i * 4096u),
                    "s"(Wh)
                 : "memory");
  }
  unsigned xdst = (unsigned)(uintptr_t)xbuf + (unsigned)tid * 8u;
  unsigned xsrc = ((unsigned)c * BTOT + (unsigned)b0) * 64u + (unsigned)tid * 8u;
  asm volatile("global_load_async_to_lds_b64 %0, %1, %2"
               :: "v"(xdst), "v"(xsrc), "s"(xh)
               : "memory");
}

// ============================================================
// Fused routing pass: recompute u_hat via WMMA, (optional) agreement+softmax,
// accumulate s[b,n,d] = sum_c coef[b,n,c] * u_hat[b,n,c,d].
// Grid: (CBLKS, BTOT/BT), 256 threads (8 waves, wave32).
// ============================================================
__global__ __launch_bounds__(256, 1)
void caps_routing_pass(const _Float16* __restrict__ Wh, const _Float16* __restrict__ xh,
                       const float* __restrict__ vin,   // cumulative v; unused if use_v==0
                       float* __restrict__ s_out, int use_v) {
  extern __shared__ char smem[];
  float* a_lds  = (float*)(smem + A_LDS_OFF);
  float* cf_lds = (float*)(smem + CF_LDS_OFF);
  float* s_lds  = (float*)(smem + S_LDS_OFF);

  const int tid   = threadIdx.x;
  const int wave  = tid >> 5;
  const int lane  = tid & 31;
  const int lhalf = lane >> 4;    // 0: lanes 0-15, 1: lanes 16-31
  const int l15   = lane & 15;
  const int jn    = wave & 1;     // which 16-batch half of the 32-batch tile
  const int mw    = wave >> 1;    // M-tile offset 0..3 (tiles mt = mw + 4t)
  const int b_loc = jn * 16 + l15;                 // local batch 0..31
  const int b0    = blockIdx.y * BT;
  const int cbase = blockIdx.x * CCHUNK;

  // zero the per-workgroup s accumulator
  for (int i = tid; i < ND * S_STRIDE; i += 256) s_lds[i] = 0.f;

  // kick off channel cbase into buffer 0
  stage_async(Wh, xh, smem + W0_OFF, smem + X0_OFF, cbase, b0, tid);

  for (int ci = 0; ci < CCHUNK; ++ci) {
    const int cur = ci & 1;
    _Float16* w_lds = (_Float16*)(smem + (cur ? W1_OFF : W0_OFF));
    _Float16* x_lds = (_Float16*)(smem + (cur ? X1_OFF : X0_OFF));

    __syncthreads();   // all waves finished reading the spare buffer (prev compute)

    if (ci + 1 < CCHUNK) {
      // stream next channel into the spare buffer while we compute this one
      stage_async(Wh, xh, smem + (cur ? W0_OFF : W1_OFF),
                  smem + (cur ? X0_OFF : X1_OFF), cbase + ci + 1, b0, tid);
      // 17 newest (next channel) may stay outstanding; 17 oldest (this channel) done
      asm volatile("s_wait_asynccnt 0x11" ::: "memory");
    } else {
      asm volatile("s_wait_asynccnt 0x0" ::: "memory");
    }
    if (use_v) for (int i = tid; i < NC * BT; i += 256) a_lds[i] = 0.f;
    __syncthreads();   // this channel's data visible to all waves

    // ---- B fragment (x, K x N = 32 x 16), constant over M-tiles ----
    // lanes 0-15: N=lane, K 0..7 & 16..23 ; lanes 16-31: N=lane-16, K 8..15 & 24..31
    union { v16h h; uint4 u[2]; } bf;
    {
      const _Float16* xp = x_lds + b_loc * DIN + lhalf * 8;
      bf.u[0] = *(const uint4*)xp; bf.u[1] = *(const uint4*)(xp + 16);
    }

    // -------- phase 1: agreement logits a[n][b] = u_hat . v --------
    if (use_v) {
      #pragma unroll 4
      for (int t = 0; t < 16; ++t) {
        const int mt = mw + 4 * t;                 // M-tile 0..63
        const int m  = mt * 16 + l15;              // A row for this lane
        union { v16h h; uint4 u[2]; } af;
        // lanes 0-15: K 0..7 & 16..23 ; lanes 16-31: K 8..15 & 24..31
        af.u[0] = *(const uint4*)(w_lds + m * DIN + lhalf * 8);
        af.u[1] = *(const uint4*)(w_lds + m * DIN + 16 + lhalf * 8);
        v8f d = {};
        d = __builtin_amdgcn_wmma_f32_16x16x32_f16(false, af.h, false, bf.h,
                                                   (short)0, d, false, false);
        const int n     = mt >> 1;
        const int dbase = (mt & 1) * 16 + lhalf * 8;   // this lane's 8 d's
        const float* vp = vin + (((size_t)(b0 + b_loc) * NC + n) * DC) + dbase;
        float acc = 0.f;
        #pragma unroll
        for (int r = 0; r < 8; ++r) acc += d[r] * vp[r];
        atomicAdd(&a_lds[n * BT + b_loc], acc);        // ds_add_f32
      }
      __syncthreads();

      // -------- phase 2: softmax over n per batch column --------
      if (tid < BT) {
        const int b = tid;
        float mx = -1e30f;
        for (int n = 0; n < NC; ++n) mx = fmaxf(mx, a_lds[n * BT + b]);
        float ssum = 0.f;
        for (int n = 0; n < NC; ++n) {
          float e = __expf(a_lds[n * BT + b] - mx);
          cf_lds[n * BT + b] = e; ssum += e;
        }
        float inv = 1.f / ssum;
        for (int n = 0; n < NC; ++n) cf_lds[n * BT + b] *= inv;
      }
      __syncthreads();
    }

    // -------- phase 3: recompute WMMA, s += coef * u_hat --------
    #pragma unroll 4
    for (int t = 0; t < 16; ++t) {
      const int mt = mw + 4 * t;
      const int m  = mt * 16 + l15;
      union { v16h h; uint4 u[2]; } af;
      af.u[0] = *(const uint4*)(w_lds + m * DIN + lhalf * 8);
      af.u[1] = *(const uint4*)(w_lds + m * DIN + 16 + lhalf * 8);
      v8f d = {};
      d = __builtin_amdgcn_wmma_f32_16x16x32_f16(false, af.h, false, bf.h,
                                                 (short)0, d, false, false);
      const int n    = mt >> 1;
      const float cf = use_v ? cf_lds[n * BT + b_loc] : (1.f / 32.f);
      const int sb   = (mt * 16 + lhalf * 8) * S_STRIDE + b_loc;
      #pragma unroll
      for (int r = 0; r < 8; ++r) s_lds[sb + r * S_STRIDE] += cf * d[r];
    }
  }

  __syncthreads();
  // ---- flush s tile to global with fp32 atomics (sum over channel blocks) ----
  for (int i = tid; i < BT * ND; i += 256) {
    const int bl = i & (BT - 1);
    const int m  = i >> 5;
    unsafeAtomicAdd(&s_out[(size_t)(b0 + bl) * ND + m], s_lds[m * S_STRIDE + bl]);
  }
}

// ============================================================
// v = squash(s) per (b,n) row of 32; optional accumulate (vsum += v)
// ============================================================
__global__ void caps_squash(const float* __restrict__ s, float* __restrict__ vout,
                            int accumulate) {
  const int idx = blockIdx.x * blockDim.x + threadIdx.x;   // (b,n), 4096 total
  if (idx >= BTOT * NC) return;
  const float* sp = s + (size_t)idx * DC;
  float vals[DC];
  float sq = 0.f;
  #pragma unroll
  for (int d = 0; d < DC; ++d) { vals[d] = sp[d]; sq += vals[d] * vals[d]; }
  const float scale = sq / (1.f + sq) / (sqrtf(sq) + 1e-8f);
  float* vp = vout + (size_t)idx * DC;
  #pragma unroll
  for (int d = 0; d < DC; ++d) {
    float v = vals[d] * scale;
    vp[d] = accumulate ? (vp[d] + v) : v;
  }
}

// ============================================================
extern "C" void kernel_launch(void* const* d_in, const int* in_sizes, int n_in,
                              void* d_out, int out_size, void* d_ws, size_t ws_size,
                              hipStream_t stream) {
  const float* x = (const float*)d_in[0];      // [128,2048,32]
  const float* W = (const float*)d_in[1];      // [1,32,2048,32,32]
  float* out = (float*)d_out;                  // [128,32,32]

  // workspace layout
  char* ws = (char*)d_ws;
  const size_t WH_BYTES = (size_t)NC * CIN * DC * DIN * 2;   // 134,217,728
  const size_t XH_BYTES = (size_t)BTOT * CIN * DIN * 2;      //  16,777,216
  const size_t S_BYTES  = (size_t)BTOT * NC * DC * 4;        //     524,288
  _Float16* Wh   = (_Float16*)ws;
  _Float16* xh   = (_Float16*)(ws + WH_BYTES);
  float*    stmp = (float*)(ws + WH_BYTES + XH_BYTES);
  float*    vsum = (float*)(ws + WH_BYTES + XH_BYTES + S_BYTES);

  // 1) one-time layout/precision conversion to f16
  {
    const int wdw = NC * CIN * DC * DIN / 2;   // 33,554,432 dwords
    caps_convert_W<<<8192, 256, 0, stream>>>(W, (unsigned*)Wh, wdw);
    const int xdw = BTOT * CIN * DIN / 2;      //  4,194,304 dwords
    caps_convert_x<<<2048, 256, 0, stream>>>(x, (unsigned*)xh, xdw);
  }

  dim3 grid(CBLKS, BTOT / BT);   // 128 x 4

  // 2) iteration 1: uniform coefficients (softmax of zeros) -> s1
  hipMemsetAsync(stmp, 0, S_BYTES, stream);
  caps_routing_pass<<<grid, 256, SMEM_BYTES, stream>>>(Wh, xh, nullptr, stmp, 0);
  caps_squash<<<16, 256, 0, stream>>>(stmp, vsum, 0);          // vsum = v1

  // 3) iteration 2: logits = u_hat . v1 -> s2
  hipMemsetAsync(stmp, 0, S_BYTES, stream);
  caps_routing_pass<<<grid, 256, SMEM_BYTES, stream>>>(Wh, xh, vsum, stmp, 1);
  caps_squash<<<16, 256, 0, stream>>>(stmp, vsum, 1);          // vsum = v1 + v2

  // 4) iteration 3: logits = u_hat . (v1+v2) -> s3 = output (no final squash)
  hipMemsetAsync(d_out, 0, S_BYTES, stream);
  caps_routing_pass<<<grid, 256, SMEM_BYTES, stream>>>(Wh, xh, vsum, out, 1);
}
